// SimpleMSA_62234076119748
// MI455X (gfx1250) — matrix operand order repaired and checked by reading, not compiled, not run on hardware
//
#include <hip/hip_runtime.h>
#include <hip/hip_bf16.h>

typedef __attribute__((ext_vector_type(16))) _Float16 v16h;
typedef __attribute__((ext_vector_type(8)))  _Float16 v8h;
typedef __attribute__((ext_vector_type(8)))  float    v8f;
typedef __attribute__((ext_vector_type(4)))  float    v4f;

#define B_   4
#define N_   4096
#define D_   64
#define TILE_J 32          // keys per mainloop iteration (K=32 for the PV WMMA)
#define NTILES (N_ / TILE_J)
#define ROWS_PER_BLOCK 128 // 8 waves x 16 query rows
#define THREADS 256

// ---------- DPP16 reduction for the |k|^2 staging (8-lane groups, all-VALU) ----------
template <int CTRL>
__device__ __forceinline__ float dppmov(float x) {
    return __builtin_bit_cast(float,
        __builtin_amdgcn_update_dpp(0, __builtin_bit_cast(int, x),
                                    CTRL, 0xF, 0xF, true));
}
__device__ __forceinline__ float redsum8(float x) {
    x += dppmov<0xB1>(x);    // quad_perm [1,0,3,2]  (xor 1)
    x += dppmov<0x4E>(x);    // quad_perm [2,3,0,1]  (xor 2)
    x += dppmov<0x141>(x);   // ROW_HALF_MIRROR      (combines quads within 8)
    return x;
}

// concatenate two 8-element f16 vectors into a 16-element WMMA operand
__device__ __forceinline__ v16h cat8(v8h a, v8h b) {
    return __builtin_shufflevector(a, b, 0, 1, 2, 3, 4, 5, 6, 7,
                                         8, 9, 10, 11, 12, 13, 14, 15);
}

// ---------- staging: f32 -> f16 row copy + transposed copy + pre-scaled |k|^2 ----------
__device__ __forceinline__ void stage_tile(_Float16* kT, _Float16* vTb, float* sqT,
                                           int jl, int d0, int tid, float invS2,
                                           v4f x0, v4f x1) {
    float ss = 0.0f;
    v8h h;
#pragma unroll
    for (int e = 0; e < 8; ++e) {
        const float v = (e < 4) ? x0[e] : x1[e - 4];
        ss += v * v;
        h[e] = (_Float16)v;
    }
    *(v8h*)&kT[jl * D_ + d0] = h;            // [key][d]  (A-frags for K*Q^T)
#pragma unroll
    for (int e = 0; e < 8; ++e)
        vTb[(d0 + e) * TILE_J + jl] = h[e];  // [d][key]  (B-frags for PV)
    ss = redsum8(ss);                        // 8 threads share a key
    if ((tid & 7) == 0) sqT[jl] = ss * invS2;   // store the ready-to-use bias
}

// Mean-squared-distance attention, flash-style:
//   t[i,j] = (2*q_i.k_j - |k_j|^2)/(D*scale)   (row-constant |q_i|^2 cancels in softmax)
// Since Q=K the exact row max is attained at j=i: M[i] = |q_i|^2/(D*scale).
// -> p = exp2(t - M) in (0,1], no online max/rescaling; l = P*ones via an extra WMMA.
// We compute S^T = K*Q^T so the exp'd probabilities land directly in the
// A-fragment register layout required by the PV WMMA (no LDS round-trip for P).
__global__ __launch_bounds__(THREADS) void msa_flash_wmma(
    const float* __restrict__ tkn, const float* __restrict__ scale_p,
    float* __restrict__ out)
{
    const int b     = blockIdx.y;
    const int qBase = blockIdx.x * ROWS_PER_BLOCK;
    const int tid   = threadIdx.x;
    const int w     = tid >> 5;     // wave id (wave32)
    const int lane  = tid & 31;
    const int lo    = lane & 15;    // position within 16-lane half
    const int hi    = lane >> 4;    // which half of the wave

    __shared__ __attribute__((aligned(32))) _Float16 kTile[2][TILE_J * D_];  // 8KB
    __shared__ __attribute__((aligned(32))) _Float16 vT[2][D_ * TILE_J];     // 8KB
    __shared__ __attribute__((aligned(32))) float    sqTile[2][TILE_J];      // 256B

    // fold log2(e) into the logit scale so native base-2 exp can be used
    const float invS2 = 1.4426950408889634f / ((float)D_ * scale_p[0]);
    const float c2    = 2.0f * invS2;
    const float* tb   = tkn + (size_t)b * N_ * D_;

    // ---- Q as WMMA B-fragments (lane = query col, element e -> d = (e<8 ? e : 8+e) + 8*hi, +32/chunk)
    //      plus this lane's |q_lo|^2 (per-lane M; no broadcast needed) ----
    v16h qb0, qb1;
    float pq = 0.0f;
    {
        const int row = qBase + w * 16 + lo;
        const float* qr = tb + (size_t)row * D_ + 8 * hi;
#pragma unroll
        for (int e = 0; e < 16; ++e) {
            const float a0 = qr[e < 8 ? e : e + 8];
            const float a1 = qr[32 + (e < 8 ? e : e + 8)];
            pq += a0 * a0 + a1 * a1;         // 32 of the row's 64 elements
            qb0[e] = (_Float16)a0;
            qb1[e] = (_Float16)a1;
        }
    }
    pq += __shfl_xor(pq, 16);                // combine the two half-wave partials
    const float M = pq * invS2;              // exact row max of the base-2 logits

    // accumulators: output (4 d-groups) + softmax denominator via P*ones
    v8f oc[4];
    v8f lsum = {0,0,0,0,0,0,0,0};
#pragma unroll
    for (int g = 0; g < 4; ++g)
#pragma unroll
        for (int r = 0; r < 8; ++r) oc[g][r] = 0.0f;

    v16h vones;
#pragma unroll
    for (int e = 0; e < 16; ++e) vones[e] = (_Float16)1.0f;

    const int jl = tid >> 3;          // local key 0..31 (8 threads per key)
    const int d0 = (tid & 7) * 8;     // 8 consecutive d per thread

    // ---- prologue: stage tile 0 into buffer 0 ----
    {
        const float* s0p = tb + (size_t)jl * D_ + d0;
        v4f x0 = *(const v4f*)(s0p);
        v4f x1 = *(const v4f*)(s0p + 4);
        stage_tile(kTile[0], vT[0], sqTile[0], jl, d0, tid, invS2, x0, x1);
    }
    __syncthreads();

    for (int tile = 0; tile < NTILES; ++tile) {
        const int  cur     = tile & 1;
        const bool hasNext = (tile + 1) < NTILES;

        // issue next tile's global loads NOW; s_wait_loadcnt deferred to stage_tile
        v4f n0 = {0,0,0,0}, n1 = {0,0,0,0};
        if (hasNext) {
            const float* srcn = tb + (size_t)((tile + 1) * TILE_J + jl) * D_ + d0;
            n0 = *(const v4f*)(srcn);
            n1 = *(const v4f*)(srcn + 4);
            if (tile + 2 < NTILES)
                __builtin_prefetch(srcn + TILE_J * D_, 0, 1);   // global_prefetch_b8
        }

        const _Float16* kT  = kTile[cur];
        const _Float16* vTb = vT[cur];

        // ---- K A-fragments: lane lo = key row (subtile s: j = s*16+lo);
        //      element e -> d = (e<8 ? e : 8+e) + 8*hi (+32/chunk): two 16B runs ----
        const int base0 = lo * D_ + 8 * hi;          // subtile 0
        const int base1 = (16 + lo) * D_ + 8 * hi;   // subtile 1
        const v16h ka00 = cat8(*(const v8h*)&kT[base0],      *(const v8h*)&kT[base0 + 16]);
        const v16h ka01 = cat8(*(const v8h*)&kT[base0 + 32], *(const v8h*)&kT[base0 + 48]);
        const v16h ka10 = cat8(*(const v8h*)&kT[base1],      *(const v8h*)&kT[base1 + 16]);
        const v16h ka11 = cat8(*(const v8h*)&kT[base1 + 32], *(const v8h*)&kT[base1 + 48]);

        // ---- S^T = K*Q^T : C layout = (lane = query col, VGPR r = key r+8*hi) ----
        v8f sT0 = {0,0,0,0,0,0,0,0}, sT1 = {0,0,0,0,0,0,0,0};
        sT0 = __builtin_amdgcn_wmma_f32_16x16x32_f16(false, ka00, false, qb0, (short)0, sT0, false, false);
        sT1 = __builtin_amdgcn_wmma_f32_16x16x32_f16(false, ka10, false, qb0, (short)0, sT1, false, false);
        sT0 = __builtin_amdgcn_wmma_f32_16x16x32_f16(false, ka01, false, qb1, (short)0, sT0, false, false);
        sT1 = __builtin_amdgcn_wmma_f32_16x16x32_f16(false, ka11, false, qb1, (short)0, sT1, false, false);

        // per-register key biases (|k_j|^2 * invS2), keys r+8*hi contiguous
        const v8f sq0v = *(const v8f*)&sqTile[cur][8 * hi];
        const v8f sq1v = *(const v8f*)&sqTile[cur][16 + 8 * hi];

        // ---- p = exp2(t - M) straight into the PV A-fragment registers ----
        v16h pa;
#pragma unroll
        for (int r = 0; r < 8; ++r) {
            const float p0 = __builtin_amdgcn_exp2f(fmaf(sT0[r], c2, -sq0v[r]) - M);
            const float p1 = __builtin_amdgcn_exp2f(fmaf(sT1[r], c2, -sq1v[r]) - M);
            pa[r]     = (_Float16)p0;   // element r    -> key r+8*hi    (subtile 0)
            pa[8 + r] = (_Float16)p1;   // element 8+r  -> key 16+r+8*hi (subtile 1)
        }

        // ---- O += P*V (4 d-groups) and l += P*ones: 5 independent accumulator chains ----
        lsum = __builtin_amdgcn_wmma_f32_16x16x32_f16(false, pa, false, vones, (short)0, lsum, false, false);
#pragma unroll
        for (int g = 0; g < 4; ++g) {
            v16h vb = cat8(*(const v8h*)&vTb[(g * 16 + lo) * TILE_J + hi * 8], *(const v8h*)&vTb[(g * 16 + lo) * TILE_J + 16 + hi * 8]);
            oc[g] = __builtin_amdgcn_wmma_f32_16x16x32_f16(false, pa, false, vb, (short)0, oc[g], false, false);
        }

        // ---- stage next tile into the other buffer; single barrier per iteration ----
        if (hasNext)
            stage_tile(kTile[1 - cur], vT[1 - cur], sqTile[1 - cur], jl, d0, tid, invS2, n0, n1);
        __syncthreads();
    }

    // ---- epilogue: normalize by the WMMA-accumulated denominator and store ----
    {
        const int rowBase = qBase + w * 16 + 8 * hi;
#pragma unroll
        for (int r = 0; r < 8; ++r) {
            const float inv = 1.0f / lsum[r];
            const size_t o = ((size_t)b * N_ + rowBase + r) * D_ + lo;
#pragma unroll
            for (int g = 0; g < 4; ++g)
                out[o + g * 16] = oc[g][r] * inv;
        }
    }
}

extern "C" void kernel_launch(void* const* d_in, const int* in_sizes, int n_in,
                              void* d_out, int out_size, void* d_ws, size_t ws_size,
                              hipStream_t stream) {
    const float* tkn   = (const float*)d_in[0];
    const float* scale = (const float*)d_in[1];
    float* out = (float*)d_out;
    dim3 grid(N_ / ROWS_PER_BLOCK, B_);   // (32, 4)
    msa_flash_wmma<<<grid, THREADS, 0, stream>>>(tkn, scale, out);
}
